// Attention_16853451669914
// MI455X (gfx1250) — matrix operand order repaired and checked by reading, not compiled, not run on hardware
//
#include <hip/hip_runtime.h>

// Attention: B=4, N=4096, D=256, fp32 in/out.
// Compute-bound (~72 GFLOP vs ~100 MB min traffic) -> flash attention on the
// f16 WMMA path (v_wmma_f32_16x16x32_f16), f32 accumulation.
// CDNA5 data path: K / V^T blocks staged into LDS once per workgroup with
// global_load_async_to_lds_b128 (ASYNCcnt), double-buffered; all 4 waves of
// the block consume them via ds_load. P tile transposed C->A layout via LDS.
// Workspace: Q,K (row-major f16) + V^T (f16) = 3 * B*N*D * 2B = 24 MB.

#define NTOK   4096
#define DMODEL 256
#define BATCH  4

typedef _Float16 v16h __attribute__((ext_vector_type(16)));
typedef _Float16 v8h  __attribute__((ext_vector_type(8)));
typedef float    v8f  __attribute__((ext_vector_type(8)));
typedef float    v4f  __attribute__((ext_vector_type(4)));

static __device__ __forceinline__ v8f wmma_f16(v16h a, v16h b, v8f c) {
  // D = A(16x32 f16) * B(32x16 f16) + C(16x16 f32)
  return __builtin_amdgcn_wmma_f32_16x16x32_f16(
      /*neg_a=*/false, a, /*neg_b=*/false, b,
      /*c_mod=*/(short)0, c, /*reuse_a=*/false, /*reuse_b=*/false);
}

// Low 32 bits of a flat pointer to __shared__ memory = LDS byte offset.
static __device__ __forceinline__ unsigned lds_off(const void* p) {
  return (unsigned)(uintptr_t)p;
}

static __device__ __forceinline__ void async_b128(unsigned lds, unsigned voff,
                                                  const void* sbase) {
  asm volatile("global_load_async_to_lds_b128 %0, %1, %2"
               :: "v"(lds), "v"(voff), "s"(sbase) : "memory");
}

// ---------------------------------------------------------------------------
// Projection: Out[n][e] = (sum_d X[n][d] * W[e][d] + bias[e]) * scale, cast f16.
// One wave per 16x16 output tile. transpose_out=1 stores Out^T ([D][N] per
// batch), used for V so the PV WMMA B-operand is a contiguous 32-byte read.
// ---------------------------------------------------------------------------
__global__ void __launch_bounds__(32) proj_f16_kernel(
    const float* __restrict__ X, const float* __restrict__ W,
    const float* __restrict__ bias, _Float16* __restrict__ Out,
    int transpose_out, float scale)
{
  const int lane = threadIdx.x & 31;
  const int m = lane & 15;        // A: row within tile / B: output column
  const int h = lane >> 4;        // lane half selects K sub-range
  const int et = blockIdx.x;      // output-feature tile (e)
  const int nt = blockIdx.y;      // token tile (n)
  const int b  = blockIdx.z;

  const float* xrow = X + ((size_t)b * NTOK + (size_t)nt * 16 + m) * DMODEL;
  const float* wrow = W + (size_t)(et * 16 + m) * DMODEL;

  v8f acc;
#pragma unroll
  for (int r = 0; r < 8; ++r) acc[r] = 0.f;

#pragma unroll
  for (int c = 0; c < 8; ++c) {
    // A-layout: lane(m,h) holds row m, K = c*32 + h*16 + 0..15
    const float* xp = xrow + c * 32 + h * 16;
    v4f a0 = *(const v4f*)(xp);
    v4f a1 = *(const v4f*)(xp + 4);
    v4f a2 = *(const v4f*)(xp + 8);
    v4f a3 = *(const v4f*)(xp + 12);
    v16h A;
#pragma unroll
    for (int i = 0; i < 4; ++i) {
      A[i]      = (_Float16)a0[i];
      A[4 + i]  = (_Float16)a1[i];
      A[8 + i]  = (_Float16)a2[i];
      A[12 + i] = (_Float16)a3[i];
    }
    // B-layout: lane(col e=et*16+m, h) holds B rows d = c*32 + h*16 + 0..15,
    // and B[d][e] = W[e][d] -> contiguous 16 floats of W row e.
    const float* wp = wrow + c * 32 + h * 16;
    v4f b0 = *(const v4f*)(wp);
    v4f b1 = *(const v4f*)(wp + 4);
    v4f b2 = *(const v4f*)(wp + 8);
    v4f b3 = *(const v4f*)(wp + 12);
    v16h Bm;
#pragma unroll
    for (int i = 0; i < 4; ++i) {
      Bm[i]      = (_Float16)b0[i];
      Bm[4 + i]  = (_Float16)b1[i];
      Bm[8 + i]  = (_Float16)b2[i];
      Bm[12 + i] = (_Float16)b3[i];
    }
    acc = wmma_f16(A, Bm, acc);
  }

  const float bv = bias[et * 16 + m];
#pragma unroll
  for (int r = 0; r < 8; ++r) acc[r] = (acc[r] + bv) * scale;

  if (transpose_out) {
    // C-layout: lane's column e is fixed, rows n = nt*16 + r + 8h are
    // consecutive -> one 16-byte store per lane into Out^T[e][n].
    v8h pk;
#pragma unroll
    for (int r = 0; r < 8; ++r) pk[r] = (_Float16)acc[r];
    _Float16* dst = Out + ((size_t)b * DMODEL + (size_t)(et * 16 + m)) * NTOK
                        + (size_t)nt * 16 + h * 8;
    *(v8h*)dst = pk;
  } else {
    _Float16* dst = Out + ((size_t)b * NTOK + (size_t)nt * 16) * DMODEL
                        + (size_t)(et * 16 + m);
#pragma unroll
    for (int r = 0; r < 8; ++r)
      dst[(size_t)(r + 8 * h) * DMODEL] = (_Float16)acc[r];
  }
}

// ---------------------------------------------------------------------------
// Flash attention: 4 waves/block, one 16-row Q tile per wave. K/V^T blocks
// (32 keys) staged to LDS via async copies, double-buffered.
// ---------------------------------------------------------------------------
__global__ void __launch_bounds__(128) attn_kernel(
    const _Float16* __restrict__ Q,   // [B][N][D] f16, pre-scaled by 1/sqrt(D)
    const _Float16* __restrict__ K,   // [B][N][D] f16
    const _Float16* __restrict__ Vt,  // [B][D][N] f16 (transposed)
    float* __restrict__ Out)          // [B][N][D] f32
{
  __shared__ __align__(16) _Float16 kbuf[2][32 * DMODEL];   // [key][d]   16KB x2
  __shared__ __align__(16) _Float16 vbuf[2][DMODEL * 32];   // [d][key]   16KB x2
  __shared__ __align__(16) _Float16 pbuf_all[4][16 * 32];   // per-wave P scratch

  const int tid  = threadIdx.x;
  const int wave = tid >> 5;
  const int lane = tid & 31;
  const int m = lane & 15;
  const int h = lane >> 4;

  const int b    = blockIdx.y;
  const int qt   = blockIdx.x * 4 + wave;
  const int qrow = qt * 16;

  const _Float16* Qb  = Q  + (size_t)b * NTOK * DMODEL;
  const _Float16* Kb  = K  + (size_t)b * NTOK * DMODEL;
  const _Float16* Vtb = Vt + (size_t)b * DMODEL * NTOK;

  const unsigned klds0 = lds_off(&kbuf[0][0]);
  const unsigned klds1 = lds_off(&kbuf[1][0]);
  const unsigned vlds0 = lds_off(&vbuf[0][0]);
  const unsigned vlds1 = lds_off(&vbuf[1][0]);

  // Cooperative async stage of one 32-key block: K rows [kb..kb+32) x D f16,
  // and V^T rows (all 256 d) x 32 keys. 16 b128 transfers per thread.
  auto stage = [&](int kb, int bufsel) {
    const unsigned kl = bufsel ? klds1 : klds0;
    const unsigned vl = bufsel ? vlds1 : vlds0;
#pragma unroll
    for (int i = 0; i < 8; ++i) {
      // K: row = i*4 + wave (0..31), 16B segment = lane (0..31)
      const unsigned krel = (unsigned)(((i * 4 + wave) * DMODEL + lane * 8) * 2);
      async_b128(kl + krel, (unsigned)(kb * DMODEL * 2) + krel, Kb);
      // V^T: d-row = i*32 + wave*8 + lane/4 (0..255), 16B segment = lane&3
      const unsigned d   = (unsigned)(i * 32 + wave * 8 + (lane >> 2));
      const unsigned seg = (unsigned)(lane & 3);
      async_b128(vl + (d * 32 + seg * 8) * 2,
                 (unsigned)((d * NTOK + (unsigned)kb + seg * 8) * 2), Vtb);
    }
  };

  // Q tile in A-layout registers: 8 chunks of K=32 (v16h each).
  v16h qa[8];
#pragma unroll
  for (int c = 0; c < 8; ++c) {
    const _Float16* qp = Qb + (size_t)(qrow + m) * DMODEL + c * 32 + h * 16;
    v8h lo = *(const v8h*)(qp);
    v8h hi = *(const v8h*)(qp + 8);
#pragma unroll
    for (int i = 0; i < 8; ++i) { qa[c][i] = lo[i]; qa[c][8 + i] = hi[i]; }
  }

  v8f o[16];                       // 16 output tiles of 16 columns (D=256)
#pragma unroll
  for (int t = 0; t < 16; ++t)
#pragma unroll
    for (int r = 0; r < 8; ++r) o[t][r] = 0.f;

  float mrow[8], lrow[8];          // per-row running max / sum (replicated
#pragma unroll                     // across each 16-lane half, matching C-layout)
  for (int r = 0; r < 8; ++r) { mrow[r] = -1e30f; lrow[r] = 0.f; }

  _Float16* pb = &pbuf_all[wave][0];

  // Prologue: fill buffer 0.
  stage(0, 0);
  asm volatile("s_wait_asynccnt 0x0" ::: "memory");
  __syncthreads();

  for (int kb = 0; kb < NTOK; kb += 32) {
    const int cur = (kb >> 5) & 1;

    // Kick off next block's staging into the other buffer.
    if (kb + 32 < NTOK) stage(kb + 32, cur ^ 1);

    // ---- S = Q * K^T for 32 keys (two 16x16 tiles), K read from LDS ----
    v8f s0, s1;
#pragma unroll
    for (int r = 0; r < 8; ++r) { s0[r] = 0.f; s1[r] = 0.f; }

    const _Float16* k0 = &kbuf[cur][m * DMODEL + h * 16];        // key j = m
    const _Float16* k1 = k0 + 16 * DMODEL;                       // key j = m+16
#pragma unroll
    for (int c = 0; c < 8; ++c) {
      v16h b0 = *(const v16h*)(k0 + c * 32);
      v16h b1 = *(const v16h*)(k1 + c * 32);
      s0 = wmma_f16(qa[c], b0, s0);
      s1 = wmma_f16(qa[c], b1, s1);
    }

    // ---- online softmax over this key block ----
    float scl[8];
#pragma unroll
    for (int r = 0; r < 8; ++r) {
      float t = fmaxf(s0[r], s1[r]);
      t = fmaxf(t, __shfl_xor(t, 1, 16));
      t = fmaxf(t, __shfl_xor(t, 2, 16));
      t = fmaxf(t, __shfl_xor(t, 4, 16));
      t = fmaxf(t, __shfl_xor(t, 8, 16));
      float mn = fmaxf(mrow[r], t);
      float sc = __expf(mrow[r] - mn);
      float p0 = __expf(s0[r] - mn);
      float p1 = __expf(s1[r] - mn);
      s0[r] = p0; s1[r] = p1;
      float rs = p0 + p1;
      rs += __shfl_xor(rs, 1, 16);
      rs += __shfl_xor(rs, 2, 16);
      rs += __shfl_xor(rs, 4, 16);
      rs += __shfl_xor(rs, 8, 16);
      mrow[r] = mn;
      lrow[r] = lrow[r] * sc + rs;
      scl[r] = sc;
    }
#pragma unroll
    for (int t = 0; t < 16; ++t)
#pragma unroll
      for (int r = 0; r < 8; ++r) o[t][r] *= scl[r];

    // ---- P: C-layout f32 -> LDS -> A-layout f16 (intra-wave only) ----
#pragma unroll
    for (int r = 0; r < 8; ++r) {
      const int M = r + 8 * h;                 // C-layout row
      pb[M * 32 + m]      = (_Float16)s0[r];   // key cols 0..15
      pb[M * 32 + 16 + m] = (_Float16)s1[r];   // key cols 16..31
    }
    asm volatile("s_wait_dscnt 0x0" ::: "memory");  // DS ops in-order per wave
    v16h pa;
    {
      v8h lo = *(const v8h*)(pb + m * 32 + h * 16);
      v8h hi = *(const v8h*)(pb + m * 32 + h * 16 + 8);
#pragma unroll
      for (int i = 0; i < 8; ++i) { pa[i] = lo[i]; pa[8 + i] = hi[i]; }
    }

    // ---- O += P * V ; V^T staging makes each B-operand a 32B LDS read ----
#pragma unroll
    for (int t = 0; t < 16; ++t) {
      v16h bv = *(const v16h*)(&vbuf[cur][(t * 16 + m) * 32 + h * 16]);
      o[t] = wmma_f16(pa, bv, o[t]);
    }

    // Own async copies done, then block-wide handshake before buffer swap.
    asm volatile("s_wait_asynccnt 0x0" ::: "memory");
    __syncthreads();
  }

  float inv[8];
#pragma unroll
  for (int r = 0; r < 8; ++r) inv[r] = 1.f / lrow[r];

  float* Ob = Out + ((size_t)b * NTOK + qrow) * DMODEL;
#pragma unroll
  for (int t = 0; t < 16; ++t) {
#pragma unroll
    for (int r = 0; r < 8; ++r) {
      const int n = r + 8 * h;
      Ob[(size_t)n * DMODEL + t * 16 + m] = o[t][r] * inv[r];
    }
  }
}

// ---------------------------------------------------------------------------
extern "C" void kernel_launch(void* const* d_in, const int* in_sizes, int n_in,
                              void* d_out, int out_size, void* d_ws, size_t ws_size,
                              hipStream_t stream) {
  const float* x    = (const float*)d_in[0];
  const float* z    = (const float*)d_in[1];
  const float* Wq_w = (const float*)d_in[2];
  const float* Wq_b = (const float*)d_in[3];
  const float* Wk_w = (const float*)d_in[4];
  const float* Wk_b = (const float*)d_in[5];
  const float* Wv_w = (const float*)d_in[6];
  const float* Wv_b = (const float*)d_in[7];

  const size_t mat = (size_t)BATCH * NTOK * DMODEL;   // elements per matrix
  _Float16* Qh  = (_Float16*)d_ws;                    // ws: 3*mat*2B = 24 MB
  _Float16* Kh  = Qh + mat;
  _Float16* Vth = Kh + mat;

  dim3 pgrid(DMODEL / 16, NTOK / 16, BATCH);
  // Fold 1/sqrt(D)=0.0625 into Q at projection time.
  proj_f16_kernel<<<pgrid, dim3(32), 0, stream>>>(x, Wq_w, Wq_b, Qh, 0, 0.0625f);
  proj_f16_kernel<<<pgrid, dim3(32), 0, stream>>>(z, Wk_w, Wk_b, Kh, 0, 1.0f);
  proj_f16_kernel<<<pgrid, dim3(32), 0, stream>>>(z, Wv_w, Wv_b, Vth, 1, 1.0f);

  dim3 agrid(NTOK / 64, BATCH);  // 4 waves/block, one 16-row Q tile per wave
  attn_kernel<<<agrid, dim3(128), 0, stream>>>(Qh, Kh, Vth, (float*)d_out);
}
